// BottleneckTransFusion_23579370455056
// MI455X (gfx1250) — matrix operand order repaired and checked by reading, not compiled, not run on hardware
//
#include <hip/hip_runtime.h>
#include <hip/hip_bf16.h>

// ---------------------------------------------------------------------------
// BottleneckTransFusion for MI455X (gfx1250).
// All GEMMs use V_WMMA_F32_16X16X32_F16 (f16 in, f32 accumulate) with
// double-buffered LDS tile staging (global latency hidden behind WMMA).
// LayerNorm / softmax / pinv-scaling / conv / residual in fp32 VALU.
// ---------------------------------------------------------------------------

#define HEADS  8
#define DH     64
#define LM     256
#define DIMD   512
#define CONVK  33
#define NP_MAX 16640   // padded length of the big layer

typedef _Float16 h16_t;
typedef __attribute__((ext_vector_type(16))) _Float16 v16h;
typedef __attribute__((ext_vector_type(8)))  _Float16 v8h;
typedef __attribute__((ext_vector_type(8)))  float    v8f;

#define LDS_STRIDE 40   // halves per row (32 + 8 pad) -> 80B row pitch, 16B aligned

// ---------------- generic batched WMMA GEMM --------------------------------
// C = beta * X + gamma * (A @ B).  Writes f32 (Cf) or f16 (Ch).
// Block = 128 threads (4 waves); block tile 32(M) x 64(N); K step 32.
// Each wave computes a 16x32 C slab = two v_wmma_f32_16x16x32_f16 per k-step.
// Double-buffered LDS pipeline: iteration t prefetches tile t+1 to registers,
// computes tile t from LDS, then stores the prefetch and barriers once.
// TRANSB=1: B given as N x K row-major; TRANSB=0: B given as K x N row-major.
// All M % 32 == 0, N tile coverage exact, K % 32 == 0 (guaranteed by host).
template <int TRANSB>
__global__ __launch_bounds__(128) void gemm_f16(
    const h16_t* __restrict__ A, const h16_t* __restrict__ B,
    float* __restrict__ Cf, h16_t* __restrict__ Ch,
    const h16_t* __restrict__ Xep,
    int K, int lda, int ldb, int ldc,
    long long sA, long long sB, long long sC, long long sX,
    float beta, float gamma)
{
  __shared__ h16_t sAt[2][32 * LDS_STRIDE];  // A tile, [m][k]
  __shared__ h16_t sBt[2][64 * LDS_STRIDE];  // B tile, [n][k] (K-contiguous)

  const int bz = blockIdx.z;
  A += (size_t)bz * sA;
  B += (size_t)bz * sB;
  if (Cf)  Cf  += (size_t)bz * sC;
  if (Ch)  Ch  += (size_t)bz * sC;
  if (Xep) Xep += (size_t)bz * sX;

  const int tid  = threadIdx.x;
  const int lane = tid & 31;
  const int wave = tid >> 5;
  const int mblk = blockIdx.y * 32;
  const int nblk = blockIdx.x * 64;
  const int wm = (wave >> 1) * 16;    // wave's M offset within block tile
  const int wn = (wave & 1) * 32;     // wave's N offset within block tile

  // staging coordinates (fixed per thread)
  const int ar   = tid >> 2;          // A row 0..31
  const int ac8  = (tid & 3) * 8;     // A col start (8 halves)
  const int bn   = tid >> 1;          // TRANSB: B row (n) 0..63
  const int bk16 = (tid & 1) * 16;    // TRANSB: k seg
  const int bkk  = tid >> 2;          // !TRANSB: B row (k) 0..31
  const int bn16 = (tid & 3) * 16;    // !TRANSB: n seg

  // fragment coordinates (ISA 7.12.2 lane layouts)
  const int fm  = wm + (lane & 15);
  const int fka = 8 * (lane >> 4);    // A: K {fka..fka+7, fka+16..fka+23}
  const int fn  = wn + (lane & 15);
  const int fkb = 8 * (lane >> 4);    // B: K {fkb..fkb+7, fkb+16..fkb+23}

  // per-thread global source pointers (advance by 32 halves per k-step)
  const h16_t* pa = A + (size_t)(mblk + ar) * lda + ac8;
  const h16_t* pb = TRANSB ? (B + (size_t)(nblk + bn) * ldb + bk16)
                           : (B + (size_t)bkk * ldb + nblk + bn16);

  // ---- prologue: tile 0 into LDS buffer 0 ----
  {
    v8h ra  = *(const v8h*)pa;
    v8h rb0 = *(const v8h*)pb;
    v8h rb1 = *(const v8h*)(pb + 8);
    *(v8h*)&sAt[0][ar * LDS_STRIDE + ac8] = ra;
    if (TRANSB) {
      *(v8h*)&sBt[0][bn * LDS_STRIDE + bk16]     = rb0;
      *(v8h*)&sBt[0][bn * LDS_STRIDE + bk16 + 8] = rb1;
    } else {
#pragma unroll
      for (int i = 0; i < 8; ++i) {
        sBt[0][(bn16 + i) * LDS_STRIDE + bkk]     = rb0[i];
        sBt[0][(bn16 + 8 + i) * LDS_STRIDE + bkk] = rb1[i];
      }
    }
  }
  __syncthreads();

  v8f acc0 = {}, acc1 = {};
  const int T = K >> 5;
  int buf = 0;

  for (int t = 0; t < T; ++t) {
    // ---- prefetch tile t+1 into registers (loads issue, no wait yet) ----
    v8h na = {}, nb0 = {}, nb1 = {};
    const bool more = (t + 1 < T);
    if (more) {
      const h16_t* qa = pa + (size_t)(t + 1) * 32;
      na = *(const v8h*)qa;
      const h16_t* qb = TRANSB ? (pb + (size_t)(t + 1) * 32)
                               : (pb + (size_t)(t + 1) * 32 * ldb);
      nb0 = *(const v8h*)qb;
      nb1 = *(const v8h*)(qb + 8);
    }

    // ---- compute tile t from LDS (16B ds_read_b128 only) ----
    v16h a, bf0, bf1;
    {
      v8h lo = *(const v8h*)&sAt[buf][fm * LDS_STRIDE + fka];
      v8h hi = *(const v8h*)&sAt[buf][fm * LDS_STRIDE + fka + 16];
#pragma unroll
      for (int i = 0; i < 8; ++i) { a[i] = lo[i]; a[i + 8] = hi[i]; }
    }
    {
      v8h lo = *(const v8h*)&sBt[buf][fn * LDS_STRIDE + fkb];
      v8h hi = *(const v8h*)&sBt[buf][fn * LDS_STRIDE + fkb + 16];
#pragma unroll
      for (int i = 0; i < 8; ++i) { bf0[i] = lo[i]; bf0[i + 8] = hi[i]; }
      v8h lo1 = *(const v8h*)&sBt[buf][(fn + 16) * LDS_STRIDE + fkb];
      v8h hi1 = *(const v8h*)&sBt[buf][(fn + 16) * LDS_STRIDE + fkb + 16];
#pragma unroll
      for (int i = 0; i < 8; ++i) { bf1[i] = lo1[i]; bf1[i + 8] = hi1[i]; }
    }

    acc0 = __builtin_amdgcn_wmma_f32_16x16x32_f16(false, a, false, bf0, (short)0, acc0, false, false);
    acc1 = __builtin_amdgcn_wmma_f32_16x16x32_f16(false, a, false, bf1, (short)0, acc1, false, false);

    // ---- store prefetched tile into the other buffer, single barrier ----
    if (more) {
      const int nb = buf ^ 1;
      *(v8h*)&sAt[nb][ar * LDS_STRIDE + ac8] = na;
      if (TRANSB) {
        *(v8h*)&sBt[nb][bn * LDS_STRIDE + bk16]     = nb0;
        *(v8h*)&sBt[nb][bn * LDS_STRIDE + bk16 + 8] = nb1;
      } else {
#pragma unroll
        for (int i = 0; i < 8; ++i) {
          sBt[nb][(bn16 + i) * LDS_STRIDE + bkk]     = nb0[i];
          sBt[nb][(bn16 + 8 + i) * LDS_STRIDE + bkk] = nb1[i];
        }
      }
      __syncthreads();
    }
    buf ^= 1;
  }

  // ---- epilogue: C = beta*X + gamma*acc ----
  const int col = lane & 15, trh = lane >> 4;
#pragma unroll
  for (int r = 0; r < 8; ++r) {
    const int row = mblk + wm + r + 8 * trh;
    const size_t i0 = (size_t)row * ldc + nblk + wn + col;
    const size_t i1 = i0 + 16;
    float v0 = gamma * acc0[r];
    float v1 = gamma * acc1[r];
    if (Xep) { v0 += beta * (float)Xep[i0]; v1 += beta * (float)Xep[i1]; }
    if (Cf) { Cf[i0] = v0;          Cf[i1] = v1; }
    else    { Ch[i0] = (h16_t)v0;   Ch[i1] = (h16_t)v1; }
  }
}

// ---------------- fp32 support kernels -------------------------------------
__global__ void cvt_f16_kernel(const float* __restrict__ in, h16_t* __restrict__ out, long long n) {
  long long i = (long long)blockIdx.x * 256 + threadIdx.x;
  if (i < n) out[i] = (h16_t)in[i];
}

// LayerNorm over concat(segA,segB) with `pad` zero rows prepended; f16 output.
__global__ __launch_bounds__(256) void ln_concat_kernel(
    const float* __restrict__ segA, int lenA, const float* __restrict__ segB,
    const float* __restrict__ w, const float* __restrict__ b,
    h16_t* __restrict__ out, int pad)
{
  const int row = blockIdx.x;
  const int tid = threadIdx.x;
  h16_t* o = out + (size_t)row * DIMD;
  if (row < pad) { o[tid] = (h16_t)0.f; o[tid + 256] = (h16_t)0.f; return; }
  const int i = row - pad;
  const float* x = (i < lenA) ? (segA + (size_t)i * DIMD) : (segB + (size_t)(i - lenA) * DIMD);
  float v0 = x[tid], v1 = x[tid + 256];
  __shared__ float red[256];
  red[tid] = v0 + v1; __syncthreads();
  for (int off = 128; off > 0; off >>= 1) { if (tid < off) red[tid] += red[tid + off]; __syncthreads(); }
  const float mu = red[0] * (1.f / DIMD);
  __syncthreads();
  const float c0 = v0 - mu, c1 = v1 - mu;
  red[tid] = c0 * c0 + c1 * c1; __syncthreads();
  for (int off = 128; off > 0; off >>= 1) { if (tid < off) red[tid] += red[tid + off]; __syncthreads(); }
  const float rs = rsqrtf(red[0] * (1.f / DIMD) + 1e-5f);
  o[tid]       = (h16_t)(c0 * rs * w[tid]       + b[tid]);
  o[tid + 256] = (h16_t)(c1 * rs * w[tid + 256] + b[tid + 256]);
}

// qkv (Np x 1536 f16) -> q,k,v in (H, Np, DH) layout; q scaled by dh^-0.5 = 0.125
__global__ void split_qkv_kernel(const h16_t* __restrict__ qkv,
                                 h16_t* __restrict__ q, h16_t* __restrict__ k, h16_t* __restrict__ v,
                                 int Np)
{
  long long idx = (long long)blockIdx.x * 256 + threadIdx.x;
  if (idx >= (long long)Np * DIMD) return;
  const int  d = (int)(idx % DIMD);
  const long long n = idx / DIMD;
  const int h = d >> 6, c = d & 63;
  const size_t dst = ((size_t)h * Np + n) * DH + c;
  const size_t src = (size_t)n * (3 * DIMD) + d;
  q[dst] = (h16_t)((float)qkv[src] * 0.125f);
  k[dst] = qkv[src + DIMD];
  v[dst] = qkv[src + 2 * DIMD];
}

// landmark means: (H, Np, DH) -> (H, LM, DH), averaging groups of l = Np/LM rows
__global__ void landmarks_kernel(const h16_t* __restrict__ t, h16_t* __restrict__ tl, int Np) {
  const int idx = blockIdx.x * 256 + threadIdx.x;           // H*LM*DH = 131072
  const int c = idx & 63, j = (idx >> 6) & 255, h = idx >> 14;
  const int l = Np >> 8;
  const h16_t* p = t + ((size_t)h * Np + (size_t)j * l) * DH + c;
  float s = 0.f;
  for (int e = 0; e < l; ++e) s += (float)p[(size_t)e * DH];
  tl[idx] = (h16_t)(s / (float)l);
}

// row softmax: f32 logits -> f16 probabilities, one block per row
__global__ __launch_bounds__(256) void softmax_rows_kernel(
    const float* __restrict__ in, h16_t* __restrict__ out, int C, long long ld)
{
  const long long row = blockIdx.x;
  const float* p = in + row * ld;
  h16_t* o = out + row * ld;
  const int tid = threadIdx.x;
  __shared__ float red[256];
  float m = -3.4e38f;
  for (int c = tid; c < C; c += 256) m = fmaxf(m, p[c]);
  red[tid] = m; __syncthreads();
  for (int off = 128; off > 0; off >>= 1) { if (tid < off) red[tid] = fmaxf(red[tid], red[tid + off]); __syncthreads(); }
  m = red[0]; __syncthreads();
  float s = 0.f;
  for (int c = tid; c < C; c += 256) s += expf(p[c] - m);
  red[tid] = s; __syncthreads();
  for (int off = 128; off > 0; off >>= 1) { if (tid < off) red[tid] += red[tid + off]; __syncthreads(); }
  const float inv = 1.f / red[0];
  for (int c = tid; c < C; c += 256) o[c] = (h16_t)(expf(p[c] - m) * inv);
}

// pinv init: scale = 1/(max_i sum_j |x| * max_j sum_i |x|); z0 = x^T * scale
__global__ __launch_bounds__(256) void pinv_init_kernel(const h16_t* __restrict__ x, h16_t* __restrict__ z) {
  const int h = blockIdx.x, t = threadIdx.x;
  const h16_t* X = x + (size_t)h * (LM * LM);
  h16_t*       Z = z + (size_t)h * (LM * LM);
  float rowsum = 0.f, colsum = 0.f;
  for (int j = 0; j < LM; ++j) {
    rowsum += fabsf((float)X[(size_t)t * LM + j]);
    colsum += fabsf((float)X[(size_t)j * LM + t]);
  }
  __shared__ float red[256];
  red[t] = rowsum; __syncthreads();
  for (int off = 128; off > 0; off >>= 1) { if (t < off) red[t] = fmaxf(red[t], red[t + off]); __syncthreads(); }
  const float cmax = red[0];        // |x|.sum(-1).max()
  __syncthreads();
  red[t] = colsum; __syncthreads();
  for (int off = 128; off > 0; off >>= 1) { if (t < off) red[t] = fmaxf(red[t], red[t + off]); __syncthreads(); }
  const float rmax = red[0];        // |x|.sum(-2).max()
  const float sc = 1.f / (cmax * rmax);
  for (int j = 0; j < LM; ++j)
    Z[(size_t)t * LM + j] = (h16_t)((float)X[(size_t)j * LM + t] * sc);
}

// depthwise conv (kernel 33 along N, per head) on v, added to attention out; f16 result
__global__ void merge_conv_kernel(const float* __restrict__ outh, const h16_t* __restrict__ v,
                                  const float* __restrict__ convw, h16_t* __restrict__ outpre, int Np)
{
  long long idx = (long long)blockIdx.x * 256 + threadIdx.x;   // Np*512
  if (idx >= (long long)Np * DIMD) return;
  const int  d = (int)(idx % DIMD);
  const long long n = idx / DIMD;
  const int h = d >> 6, c = d & 63;
  const h16_t* vp = v + ((size_t)h * Np) * DH + c;
  float acc = outh[((size_t)h * Np + n) * DH + c];
  for (int t = 0; t < CONVK; ++t) {
    const long long nn = n + t - (CONVK / 2);
    if (nn >= 0 && nn < Np) acc += (float)vp[(size_t)nn * DH] * convw[h * CONVK + t];
  }
  outpre[(size_t)n * DIMD + d] = (h16_t)acc;
}

// out[i] = x_cat[i]*omega + proj[pad+i] + bias  (drops the `pad` front rows)
__global__ void final_residual_kernel(const float* __restrict__ proj, const float* __restrict__ bias,
                                      const float* __restrict__ segA, int lenA,
                                      const float* __restrict__ segB,
                                      const float* __restrict__ omega,
                                      float* __restrict__ out, int pad, int Nout)
{
  long long idx = (long long)blockIdx.x * 256 + threadIdx.x;
  if (idx >= (long long)Nout * DIMD) return;
  const int  d = (int)(idx % DIMD);
  const long long i = idx / DIMD;
  const float* x = (i < lenA) ? (segA + (size_t)i * DIMD) : (segB + (size_t)(i - lenA) * DIMD);
  out[idx] = x[d] * omega[0] + proj[((size_t)(pad + i)) * DIMD + d] + bias[d];
}

// ---------------- host-side orchestration ----------------------------------
static inline void launch_gemm(hipStream_t s,
    const h16_t* A, const h16_t* B, float* Cf, h16_t* Ch, const h16_t* X,
    int M, int N, int K, int lda, int ldb, int ldc,
    long long sA, long long sB, long long sC, long long sX,
    int transB, float beta, float gamma, int batch)
{
  dim3 g((unsigned)((N + 63) / 64), (unsigned)((M + 31) / 32), (unsigned)batch);
  if (transB)
    gemm_f16<1><<<g, 128, 0, s>>>(A, B, Cf, Ch, X, K, lda, ldb, ldc, sA, sB, sC, sX, beta, gamma);
  else
    gemm_f16<0><<<g, 128, 0, s>>>(A, B, Cf, Ch, X, K, lda, ldb, ldc, sA, sB, sC, sX, beta, gamma);
}

struct Bufs {
  h16_t *h16, *qkv16, *q16, *k16, *v16, *ql16, *kl16, *attn1, *a3w;
  h16_t *x16, *zA, *zB, *xz, *t0, *t1, *kv;
  float *logf, *log2f, *outh, *proj;
};

static void run_layer(hipStream_t s, const Bufs& B,
                      const float* segA, int lenA, const float* segB, int lenB,
                      const float* ln_w, const float* ln_b,
                      const h16_t* qkvw16, const h16_t* outw16,
                      const float* out_b, const float* conv_w, const float* omega,
                      float* outbuf, int Nout)
{
  const int N   = lenA + lenB;
  const int pad = (LM - (N % LM)) % LM;
  const int Np  = N + pad;
  const long long NpLL = Np;

  // 1) layernorm + concat + front zero-pad -> h16 (Np x 512)
  ln_concat_kernel<<<Np, 256, 0, s>>>(segA, lenA, segB, ln_w, ln_b, B.h16, pad);

  // 2) qkv = h @ Wqkv^T  (Np x 1536)
  launch_gemm(s, B.h16, qkvw16, nullptr, B.qkv16, nullptr,
              Np, 3 * DIMD, DIMD, DIMD, DIMD, 3 * DIMD, 0, 0, 0, 0, 1, 0.f, 1.f, 1);

  // 3) split heads (+ q * dh^-0.5)
  {
    long long tot = (long long)Np * DIMD;
    split_qkv_kernel<<<(unsigned)((tot + 255) / 256), 256, 0, s>>>(B.qkv16, B.q16, B.k16, B.v16, Np);
  }

  // 4) landmark means
  landmarks_kernel<<<(HEADS * LM * DH) / 256, 256, 0, s>>>(B.q16, B.ql16, Np);
  landmarks_kernel<<<(HEADS * LM * DH) / 256, 256, 0, s>>>(B.k16, B.kl16, Np);

  // 5) attn1 = softmax(q @ kl^T)   (H, Np, LM)
  launch_gemm(s, B.q16, B.kl16, B.logf, nullptr, nullptr,
              Np, LM, DH, DH, DH, LM, NpLL * DH, LM * DH, NpLL * LM, 0, 1, 0.f, 1.f, HEADS);
  softmax_rows_kernel<<<(unsigned)(HEADS * NpLL), 256, 0, s>>>(B.logf, B.attn1, LM, LM);

  // 6) attn2 = softmax(ql @ kl^T)  (H, LM, LM)
  launch_gemm(s, B.ql16, B.kl16, B.log2f, nullptr, nullptr,
              LM, LM, DH, DH, DH, LM, LM * DH, LM * DH, LM * LM, 0, 1, 0.f, 1.f, HEADS);
  softmax_rows_kernel<<<HEADS * LM, 256, 0, s>>>(B.log2f, B.x16, LM, LM);

  // 7) Moore-Penrose pinv via Newton-Schulz (f16 WMMA, fused beta*X+gamma*acc)
  pinv_init_kernel<<<HEADS, 256, 0, s>>>(B.x16, B.zA);
  h16_t* z  = B.zA;
  h16_t* zn = B.zB;
  const long long SQ = LM * LM;
  for (int it = 0; it < 6; ++it) {
    // xz = x @ z
    launch_gemm(s, B.x16, z, nullptr, B.xz, nullptr,
                LM, LM, LM, LM, LM, LM, SQ, SQ, SQ, 0, 0, 0.f, 1.f, HEADS);
    // t0 = xz @ (7I - xz) = 7*xz - xz@xz
    launch_gemm(s, B.xz, B.xz, nullptr, B.t0, B.xz,
                LM, LM, LM, LM, LM, LM, SQ, SQ, SQ, SQ, 0, 7.f, -1.f, HEADS);
    // t1 = xz @ (15I - t0) = 15*xz - xz@t0
    launch_gemm(s, B.xz, B.t0, nullptr, B.t1, B.xz,
                LM, LM, LM, LM, LM, LM, SQ, SQ, SQ, SQ, 0, 15.f, -1.f, HEADS);
    // z' = 0.25 * z @ (13I - t1) = 3.25*z - 0.25*z@t1
    launch_gemm(s, z, B.t1, nullptr, zn, z,
                LM, LM, LM, LM, LM, LM, SQ, SQ, SQ, SQ, 0, 3.25f, -0.25f, HEADS);
    h16_t* tmp = z; z = zn; zn = tmp;
  }

  // 8) attn3 = softmax(ql @ k^T)   (H, LM, Np)
  launch_gemm(s, B.ql16, B.k16, B.logf, nullptr, nullptr,
              LM, Np, DH, DH, DH, Np, LM * DH, NpLL * DH, LM * NpLL, 0, 1, 0.f, 1.f, HEADS);
  softmax_rows_kernel<<<HEADS * LM, 256, 0, s>>>(B.logf, B.a3w, Np, NpLL);

  // 9) kv = attn3 @ v              (H, LM, DH)
  launch_gemm(s, B.a3w, B.v16, nullptr, B.kv, nullptr,
              LM, DH, Np, Np, DH, DH, LM * NpLL, NpLL * DH, LM * DH, 0, 0, 0.f, 1.f, HEADS);

  // 10) w = attn1 @ z  (reuse attn3 buffer; attn3 consumed)
  launch_gemm(s, B.attn1, z, nullptr, B.a3w, nullptr,
              Np, LM, LM, LM, LM, LM, NpLL * LM, SQ, NpLL * LM, 0, 0, 0.f, 1.f, HEADS);

  // 11) outh = w @ kv              (H, Np, DH) fp32 (into logf region)
  launch_gemm(s, B.a3w, B.kv, B.outh, nullptr, nullptr,
              Np, DH, LM, LM, DH, DH, NpLL * LM, LM * DH, NpLL * DH, 0, 0, 0.f, 1.f, HEADS);

  // 12) depthwise conv residual + head merge -> outpre16 (reuse h16)
  {
    long long tot = (long long)Np * DIMD;
    merge_conv_kernel<<<(unsigned)((tot + 255) / 256), 256, 0, s>>>(B.outh, B.v16, conv_w, B.h16, Np);
  }

  // 13) proj = outpre @ Wout^T     (Np x 512 fp32)
  launch_gemm(s, B.h16, outw16, B.proj, nullptr, nullptr,
              Np, DIMD, DIMD, DIMD, DIMD, DIMD, 0, 0, 0, 0, 1, 0.f, 1.f, 1);

  // 14) out = x*omega + proj[pad:] + bias  (keep first Nout rows)
  {
    long long tot = (long long)Nout * DIMD;
    final_residual_kernel<<<(unsigned)((tot + 255) / 256), 256, 0, s>>>(
        B.proj, out_b, segA, lenA, segB, omega, outbuf, pad, Nout);
  }
}

extern "C" void kernel_launch(void* const* d_in, const int* in_sizes, int n_in,
                              void* d_out, int out_size, void* d_ws, size_t ws_size,
                              hipStream_t stream)
{
  (void)in_sizes; (void)n_in; (void)out_size; (void)ws_size;
  const float* x1      = (const float*)d_in[0];
  const float* x2      = (const float*)d_in[1];
  const float* bneck   = (const float*)d_in[2];
  const float* ln1_w   = (const float*)d_in[3];
  const float* ln1_b   = (const float*)d_in[4];
  const float* qkv1_w  = (const float*)d_in[5];
  const float* out1_w  = (const float*)d_in[6];
  const float* out1_b  = (const float*)d_in[7];
  const float* conv1_w = (const float*)d_in[8];
  const float* omega1  = (const float*)d_in[9];
  const float* ln2_w   = (const float*)d_in[10];
  const float* ln2_b   = (const float*)d_in[11];
  const float* qkv2_w  = (const float*)d_in[12];
  const float* out2_w  = (const float*)d_in[13];
  const float* out2_b  = (const float*)d_in[14];
  const float* conv2_w = (const float*)d_in[15];
  const float* omega2  = (const float*)d_in[16];

  // ---------------- workspace arena (~405 MB, sized for Np_max) ----------------
  char* p = (char*)d_ws;
  auto alloc = [&](size_t bytes) -> char* {
    char* r = p; p += (bytes + 255) & ~(size_t)255; return r;
  };
  const long long NpM = NP_MAX;

  h16_t* qkvw1_16 = (h16_t*)alloc((size_t)3 * DIMD * DIMD * 2);
  h16_t* outw1_16 = (h16_t*)alloc((size_t)DIMD * DIMD * 2);
  h16_t* qkvw2_16 = (h16_t*)alloc((size_t)3 * DIMD * DIMD * 2);
  h16_t* outw2_16 = (h16_t*)alloc((size_t)DIMD * DIMD * 2);
  float* l2out    = (float*)alloc((size_t)2052 * DIMD * 4);

  Bufs B;
  B.h16   = (h16_t*)alloc((size_t)NpM * DIMD * 2);            // LN out; later outpre
  B.qkv16 = (h16_t*)alloc((size_t)NpM * 3 * DIMD * 2);
  B.q16   = (h16_t*)alloc((size_t)HEADS * NpM * DH * 2);
  B.k16   = (h16_t*)alloc((size_t)HEADS * NpM * DH * 2);
  B.v16   = (h16_t*)alloc((size_t)HEADS * NpM * DH * 2);
  B.ql16  = (h16_t*)alloc((size_t)HEADS * LM * DH * 2);
  B.kl16  = (h16_t*)alloc((size_t)HEADS * LM * DH * 2);
  char* logregion = alloc((size_t)HEADS * NpM * LM * 4);      // logits; later outh+proj
  B.logf  = (float*)logregion;
  B.outh  = (float*)logregion;                                 // HEADS*NpM*DH f32
  B.proj  = (float*)(logregion + (size_t)HEADS * NpM * DH * 4);// NpM*DIMD f32
  B.log2f = (float*)alloc((size_t)HEADS * LM * LM * 4);
  B.attn1 = (h16_t*)alloc((size_t)HEADS * NpM * LM * 2);
  B.a3w   = (h16_t*)alloc((size_t)HEADS * NpM * LM * 2);       // attn3; later attn1@z
  B.x16   = (h16_t*)alloc((size_t)HEADS * LM * LM * 2);
  B.zA    = (h16_t*)alloc((size_t)HEADS * LM * LM * 2);
  B.zB    = (h16_t*)alloc((size_t)HEADS * LM * LM * 2);
  B.xz    = (h16_t*)alloc((size_t)HEADS * LM * LM * 2);
  B.t0    = (h16_t*)alloc((size_t)HEADS * LM * LM * 2);
  B.t1    = (h16_t*)alloc((size_t)HEADS * LM * LM * 2);
  B.kv    = (h16_t*)alloc((size_t)HEADS * LM * DH * 2);

  // ---------------- weight conversion f32 -> f16 ----------------
  {
    const long long nw = (long long)3 * DIMD * DIMD;
    cvt_f16_kernel<<<(unsigned)((nw + 255) / 256), 256, 0, stream>>>(qkv1_w, qkvw1_16, nw);
    cvt_f16_kernel<<<(unsigned)((nw + 255) / 256), 256, 0, stream>>>(qkv2_w, qkvw2_16, nw);
    const long long no = (long long)DIMD * DIMD;
    cvt_f16_kernel<<<(unsigned)((no + 255) / 256), 256, 0, stream>>>(out1_w, outw1_16, no);
    cvt_f16_kernel<<<(unsigned)((no + 255) / 256), 256, 0, stream>>>(out2_w, outw2_16, no);
  }

  // Layer 2 on [bottleneck; x2] -> l2out (2052 x 512); bn = first 4 rows
  run_layer(stream, B, bneck, 4, x2, 2048,
            ln2_w, ln2_b, qkvw2_16, outw2_16, out2_b, conv2_w, omega2,
            l2out, 2052);

  // Layer 1 on [x1; bn] -> d_out (first 16384 rows)
  run_layer(stream, B, x1, 16384, l2out, 4,
            ln1_w, ln1_b, qkvw1_16, outw1_16, out1_b, conv1_w, omega1,
            (float*)d_out, 16384);
}
